// ContextualMURefinement_7206955123417
// MI455X (gfx1250) — matrix operand order repaired and checked by reading, not compiled, not run on hardware
//
#include <hip/hip_runtime.h>

// Harness contract: kernel_launch only, all kernels on `stream`, deterministic,
// no alloc/sync during graph capture.

#define USE_TDM_PATH 1

#define B_   64
#define T_   8192
#define D_   64
#define H_   4
#define HD_  16
#define FF_  128
#define TT   32          // tokens per chunk (2 WMMA M-tiles)
#define TE2  (TT + 2)    // with +/-1 halo
#define QKVW 192         // unified q|k|v row width
#define BPB  32          // blocks per batch row
#define NCHUNK (T_ / TT) // 256 chunks per batch row

typedef _Float16 h16 __attribute__((ext_vector_type(16)));
typedef _Float16 h8  __attribute__((ext_vector_type(8)));
typedef float    f8  __attribute__((ext_vector_type(8)));
typedef unsigned int u32x4 __attribute__((ext_vector_type(4)));
typedef int i32x8 __attribute__((ext_vector_type(8)));
typedef int i32x4 __attribute__((ext_vector_type(4)));

// A fragment: 16 (rows=tokens) x 32 (K) f16, ISA layout:
//  lanes 0-15 : M=lane, K = k0 + {0..7, 16..23}
//  lanes 16-31: M=lane-16, K = k0 + {8..15, 24..31}
__device__ __forceinline__ h16 load_fragA(const _Float16* buf, int row0, int rs, int k0) {
  const int lane = threadIdx.x & 31;
  const _Float16* p = buf + (row0 + (lane & 15)) * rs + k0 + ((lane & 16) ? 8 : 0);
  union { h16 v; h8 h[2]; } u;
  u.h[0] = *(const h8*)(p);
  u.h[1] = *(const h8*)(p + 16);
  return u.v;
}

// B fragment: 32 (K) x 16 (N) f16; weights stored row-major [n][k] (x @ W^T).
// ISA layout: lanes 0-15 hold K = k0+{0..7,16..23} (N=lane), lanes 16-31 hold
// K = k0+{8..15,24..31} (N=lane-16) -> same K order as fragA.
__device__ __forceinline__ h16 load_fragB(const _Float16* w, int n0, int ldk, int k0) {
  const int lane = threadIdx.x & 31;
  const _Float16* p = w + (n0 + (lane & 15)) * ldk + k0 + ((lane & 16) ? 8 : 0);
  union { h16 v; h8 h[2]; } u;
  u.h[0] = *(const h8*)(p);
  u.h[1] = *(const h8*)(p + 16);
  return u.v;
}

#define WMMA_F16(a, bfr, c) \
  __builtin_amdgcn_wmma_f32_16x16x32_f16(false, (a), false, (bfr), (short)0, (c), false, false)

__device__ __forceinline__ float red8(float v) {   // sum over 8-lane group
  v += __shfl_xor(v, 1, 8);
  v += __shfl_xor(v, 2, 8);
  v += __shfl_xor(v, 4, 8);
  return v;
}

__global__ __launch_bounds__(256)
void mu_refine_kernel(const float* __restrict__ mu,
                      const float* __restrict__ in_w,  const float* __restrict__ in_b,
                      const float* __restrict__ ow,    const float* __restrict__ ob,
                      const float* __restrict__ ln1w,  const float* __restrict__ ln1b,
                      const float* __restrict__ ln2w,  const float* __restrict__ ln2b,
                      const float* __restrict__ l1w,   const float* __restrict__ l1b,
                      const float* __restrict__ l2w,   const float* __restrict__ l2b,
                      const float* __restrict__ gw,    const float* __restrict__ gb,
                      float* __restrict__ out) {
  // ---- LDS (~148 KB): f16 weights + biases + double-buffered mu + activations ----
  __shared__ __align__(16) _Float16 sWin [192 * 64];   // q|k|v rows
  __shared__ __align__(16) _Float16 sWout[ 64 * 64];
  __shared__ __align__(16) _Float16 sWl1 [128 * 64];
  __shared__ __align__(16) _Float16 sWl2 [ 64 * 128];
  __shared__ __align__(16) _Float16 sWg  [ 64 * 128];
  __shared__ float sBias[512];   // [0,192) inp_b | [192,256) out_b | [256,384) l1b | [384,448) l2b | [448,512) gb
  __shared__ float sLn[256];     // ln1w ln1b ln2w ln2b
  __shared__ __align__(16) float    sMu [2][TE2 * D_]; // double buffer; halo rows 0 & 33
  __shared__ __align__(16) _Float16 sMuH[TT  * D_];    // f16 mu interior (gate A)
  __shared__ __align__(16) _Float16 sXn [TE2 * D_];
  __shared__ __align__(16) _Float16 sQKV[TE2 * QKVW];  // [row][q0..63|k64..127|v128..191]
  __shared__ __align__(16) _Float16 sCtx[TT  * D_];
  __shared__ __align__(16) float    sX  [TT  * D_];    // x, then refined (f32)
  __shared__ __align__(16) _Float16 sY2 [TT  * D_];    // xn2, then refined (f16)
  __shared__ __align__(16) _Float16 sH  [TT  * FF_];

  const int tid  = threadIdx.x;
  const int lane = tid & 31;
  const int wid  = tid >> 5;
  const int b    = blockIdx.x / BPB;
  const int cb   = blockIdx.x % BPB;

  // ---- one-time weight/bias preload (f32 -> f16) ----
  for (int i = tid; i < 192 * 64; i += 256) sWin[i]  = (_Float16)in_w[i];
  for (int i = tid; i <  64 * 64; i += 256) sWout[i] = (_Float16)ow[i];
  for (int i = tid; i < 128 * 64; i += 256) sWl1[i]  = (_Float16)l1w[i];
  for (int i = tid; i < 64 * 128; i += 256) sWl2[i]  = (_Float16)l2w[i];
  for (int i = tid; i < 64 * 128; i += 256) sWg[i]   = (_Float16)gw[i];
  if (tid < 192) sBias[tid]        = in_b[tid];
  if (tid <  64) sBias[192 + tid]  = ob[tid];
  if (tid < 128) sBias[256 + tid]  = l1b[tid];
  if (tid <  64) sBias[384 + tid]  = l2b[tid];
  if (tid <  64) sBias[448 + tid]  = gb[tid];
  if (tid <  64) { sLn[tid] = ln1w[tid]; sLn[64+tid] = ln1b[tid];
                   sLn[128+tid] = ln2w[tid]; sLn[192+tid] = ln2b[tid]; }

  // ---- mu tile loader: TDM (async) for interior rows 1..TT + VMEM halo ----
  auto issue_mu_load = [&](int buf, int t0) {
#if USE_TDM_PATH && __has_builtin(__builtin_amdgcn_tensor_load_to_lds)
    if (wid == 0) {
      unsigned long long ga = (unsigned long long)(const void*)(mu + ((size_t)b * T_ + t0) * D_);
      unsigned int ldsoff = (unsigned int)(size_t)(void*)(&sMu[buf][D_]);
      u32x4 g0 = { 1u,                                  // count=1 (valid D#)
                   ldsoff,                              // lds_addr
                   (unsigned int)(ga & 0xffffffffull),  // global_addr[31:0]
                   (unsigned int)((ga >> 32) & 0x1ffffffull) | (2u << 30) }; // [56:32] | type=2
      i32x8 g1 = { (int)(2u << 16),   // data_size = 4B
                   (int)(D_ << 16),   // tensor_dim0 = 64
                   (int)(TT << 16),   // tensor_dim1 = 32
                   (int)(D_ << 16),   // tile_dim0   = 64
                   (int)TT,           // tile_dim1   = 32 (tile_dim2 = 0)
                   (int)D_,           // tensor_dim0_stride = 64
                   0, 0 };
      i32x4 gz4 = { 0, 0, 0, 0 };
      i32x8 gz8 = { 0, 0, 0, 0, 0, 0, 0, 0 };
      __builtin_amdgcn_tensor_load_to_lds(g0, g1, gz4, gz4, gz8, 0);
    }
#else
    for (int i = tid; i < TT * D_; i += 256)
      sMu[buf][D_ + i] = mu[((size_t)b * T_ + t0) * D_ + i];
#endif
    if (tid < 2 * D_) {          // halo rows 0 (t0-1) and TT+1 (t0+TT); OOB -> 0
      int side = tid >> 6, j = tid & 63;
      int r  = side ? (TT + 1) : 0;
      int gt = t0 - 1 + side * (TT + 1);
      float v = 0.f;
      if (gt >= 0 && gt < T_) v = mu[((size_t)b * T_ + gt) * D_ + j];
      sMu[buf][r * D_ + j] = v;
    }
  };

  int pb = 0;
  issue_mu_load(0, cb * TT);     // prologue: chunk 0 in flight

  for (int chunk = cb; chunk < NCHUNK; chunk += BPB) {
    const int t0 = chunk * TT;
    const int nb = pb ^ 1;
    const bool pref = (chunk + BPB) < NCHUNK;
    if (pref) issue_mu_load(nb, (chunk + BPB) * TT);   // async prefetch next chunk
#if USE_TDM_PATH && __has_builtin(__builtin_amdgcn_tensor_load_to_lds)
    if (wid == 0) {              // TDM in-order per wave: <=1 leaves current done
      if (pref) __builtin_amdgcn_s_wait_tensorcnt(1);
      else      __builtin_amdgcn_s_wait_tensorcnt(0);
    }
#endif
    __syncthreads();             // sMu[pb] (and, 1st iter, weights) now valid
    const float* sMuC = &sMu[pb][0];

    // ============ LN1 (34 rows, 8 lanes/row) + muH f16 ============
    {
      const int sub = tid & 7;
      for (int r = tid >> 3; r < TE2; r += 32) {
        const int gt = t0 - 1 + r;
        const float* row = &sMuC[r * D_];
        const float4 a = *(const float4*)(row + sub * 8);
        const float4 c = *(const float4*)(row + sub * 8 + 4);
        const float e[8] = {a.x, a.y, a.z, a.w, c.x, c.y, c.z, c.w};
        float s = e[0]+e[1]+e[2]+e[3]+e[4]+e[5]+e[6]+e[7];
        const float m = red8(s) * (1.0f / 64.0f);
        float vs = 0.f;
        for (int i = 0; i < 8; ++i) { float d = e[i] - m; vs += d * d; }
        const float inv = rsqrtf(red8(vs) * (1.0f / 64.0f) + 1e-5f);
        const bool ok = (gt >= 0 && gt < T_);
        h8 hx, hm;
        for (int i = 0; i < 8; ++i) {
          int j = sub * 8 + i;
          hx[i] = (_Float16)(ok ? ((e[i] - m) * inv * sLn[j] + sLn[64 + j]) : 0.f);
          hm[i] = (_Float16)e[i];
        }
        *(h8*)(&sXn[r * D_ + sub * 8]) = hx;
        if (r >= 1 && r <= TT) *(h8*)(&sMuH[(r - 1) * D_ + sub * 8]) = hm;
      }
    }
    __syncthreads();

    // ============ QKV GEMM: 192 outs x 32 tokens (24 wmma tile-jobs) ============
    for (int job = wid; job < 24; job += 8) {
      const int nt = job % 12, mt = job / 12;
      f8 acc = {};
      for (int k0 = 0; k0 < 64; k0 += 32) {
        h16 a  = load_fragA(sXn + D_, mt * 16, D_, k0);   // interior tokens
        h16 bf = load_fragB(sWin, nt * 16, 64, k0);
        acc = WMMA_F16(a, bf, acc);
      }
      const int mo = (lane & 16) ? 8 : 0;
      const int n  = nt * 16 + (lane & 15);
      const float bias = sBias[n];
      for (int r = 0; r < 8; ++r) {
        int t = mt * 16 + r + mo;
        sQKV[(t + 1) * QKVW + n] = (_Float16)(acc[r] + bias);   // branch-free store
      }
    }
    // halo k/v (2 tokens x 128 outputs), scalar dot64
    {
      const int o = tid & 127, side = tid >> 7;
      const int hr = side ? (TT + 1) : 0;
      const _Float16* xr = &sXn[hr * D_];
      const _Float16* wr = &sWin[(64 + o) * 64];
      float acc = sBias[64 + o];
      for (int j = 0; j < 64; ++j) acc += (float)xr[j] * (float)wr[j];
      sQKV[hr * QKVW + 64 + o] = (_Float16)acc;
    }
    __syncthreads();

    // ============ window-3 attention (128 token-head jobs) ============
    if (tid < TT * H_) {
      const int t = tid >> 2, h = tid & 3;
      float q[16];
      for (int d = 0; d < 16; ++d) q[d] = (float)sQKV[(t + 1) * QKVW + h * 16 + d];
      float sc[3];
      for (int w = 0; w < 3; ++w) {
        const int kr = t + w, gt = t0 + t - 1 + w;
        float s = 0.f;
        for (int d = 0; d < 16; ++d) s += q[d] * (float)sQKV[kr * QKVW + 64 + h * 16 + d];
        sc[w] = (gt >= 0 && gt < T_) ? s * 0.25f : -1e9f;
      }
      float m = fmaxf(sc[0], fmaxf(sc[1], sc[2]));
      float e0 = __expf(sc[0] - m), e1 = __expf(sc[1] - m), e2 = __expf(sc[2] - m);
      float inv = 1.0f / (e0 + e1 + e2);
      e0 *= inv; e1 *= inv; e2 *= inv;
      for (int d = 0; d < 16; ++d) {
        const int vc = 128 + h * 16 + d;
        float c = e0 * (float)sQKV[t * QKVW + vc]
                + e1 * (float)sQKV[(t + 1) * QKVW + vc]
                + e2 * (float)sQKV[(t + 2) * QKVW + vc];
        sCtx[t * D_ + h * 16 + d] = (_Float16)c;
      }
    }
    __syncthreads();

    // ============ out-proj + residual: x = mu + ctx @ Wout^T + b ============
    if (wid < 8) {
      const int job = wid, nt = job & 3, mt = job >> 2;
      f8 acc = {};
      for (int k0 = 0; k0 < 64; k0 += 32) {
        h16 a  = load_fragA(sCtx, mt * 16, D_, k0);
        h16 bf = load_fragB(sWout, nt * 16, 64, k0);
        acc = WMMA_F16(a, bf, acc);
      }
      const int mo = (lane & 16) ? 8 : 0;
      const int n  = nt * 16 + (lane & 15);
      const float bias = sBias[192 + n];
      for (int r = 0; r < 8; ++r) {
        int t = mt * 16 + r + mo;
        sX[t * D_ + n] = acc[r] + bias + sMuC[(t + 1) * D_ + n];
      }
    }
    __syncthreads();

    // ============ LN2 -> xn2 (f16), 8 lanes/row ============
    {
      const int sub = tid & 7, r = tid >> 3;   // r in 0..31
      const float* row = &sX[r * D_];
      const float4 a = *(const float4*)(row + sub * 8);
      const float4 c = *(const float4*)(row + sub * 8 + 4);
      const float e[8] = {a.x, a.y, a.z, a.w, c.x, c.y, c.z, c.w};
      float s = e[0]+e[1]+e[2]+e[3]+e[4]+e[5]+e[6]+e[7];
      const float m = red8(s) * (1.0f / 64.0f);
      float vs = 0.f;
      for (int i = 0; i < 8; ++i) { float d = e[i] - m; vs += d * d; }
      const float inv = rsqrtf(red8(vs) * (1.0f / 64.0f) + 1e-5f);
      h8 hx;
      for (int i = 0; i < 8; ++i) {
        int j = sub * 8 + i;
        hx[i] = (_Float16)((e[i] - m) * inv * sLn[128 + j] + sLn[192 + j]);
      }
      *(h8*)(&sY2[r * D_ + sub * 8]) = hx;
    }
    __syncthreads();

    // ============ FF1: h = relu(xn2 @ l1w^T + b) (16 jobs) ============
    for (int job = wid; job < 16; job += 8) {
      const int nt = job & 7, mt = job >> 3;
      f8 acc = {};
      for (int k0 = 0; k0 < 64; k0 += 32) {
        h16 a  = load_fragA(sY2, mt * 16, D_, k0);
        h16 bf = load_fragB(sWl1, nt * 16, 64, k0);
        acc = WMMA_F16(a, bf, acc);
      }
      const int mo = (lane & 16) ? 8 : 0;
      const int n  = nt * 16 + (lane & 15);
      const float bias = sBias[256 + n];
      for (int r = 0; r < 8; ++r) {
        int t = mt * 16 + r + mo;
        sH[t * FF_ + n] = (_Float16)fmaxf(acc[r] + bias, 0.f);
      }
    }
    __syncthreads();

    // ============ FF2 + residual: refined = x + h @ l2w^T + b ============
    if (wid < 8) {
      const int job = wid, nt = job & 3, mt = job >> 2;
      f8 acc = {};
      for (int k0 = 0; k0 < 128; k0 += 32) {
        h16 a  = load_fragA(sH, mt * 16, FF_, k0);
        h16 bf = load_fragB(sWl2, nt * 16, 128, k0);
        acc = WMMA_F16(a, bf, acc);
      }
      const int mo = (lane & 16) ? 8 : 0;
      const int n  = nt * 16 + (lane & 15);
      const float bias = sBias[384 + n];
      for (int r = 0; r < 8; ++r) {
        int t = mt * 16 + r + mo;
        float ref = acc[r] + bias + sX[t * D_ + n];
        sX[t * D_ + n]  = ref;            // refined f32
        sY2[t * D_ + n] = (_Float16)ref;  // refined f16 (gate A, reuses xn2 buf)
      }
    }
    __syncthreads();

    // ============ gate = sigmoid([mu|refined] @ gw^T + b); write out ============
    if (wid < 8) {
      const int job = wid, nt = job & 3, mt = job >> 2;
      f8 acc = {};
      for (int k0 = 0; k0 < 128; k0 += 32) {
        h16 a = (k0 < 64) ? load_fragA(sMuH, mt * 16, D_, k0)
                          : load_fragA(sY2,  mt * 16, D_, k0 - 64);
        h16 bf = load_fragB(sWg, nt * 16, 128, k0);
        acc = WMMA_F16(a, bf, acc);
      }
      const int mo = (lane & 16) ? 8 : 0;
      const int n  = nt * 16 + (lane & 15);
      const float bias = sBias[448 + n];
      for (int r = 0; r < 8; ++r) {
        int t = mt * 16 + r + mo;
        float g   = 1.0f / (1.0f + __expf(-(acc[r] + bias)));
        float ref = sX[t * D_ + n];
        float muv = sMuC[(t + 1) * D_ + n];
        out[((size_t)b * T_ + (t0 + t)) * D_ + n] = g * ref + (1.0f - g) * muv;
      }
    }
    __syncthreads();   // buffers reused next chunk
    pb = nb;
  }
}

extern "C" void kernel_launch(void* const* d_in, const int* in_sizes, int n_in,
                              void* d_out, int out_size, void* d_ws, size_t ws_size,
                              hipStream_t stream) {
  const float* mu   = (const float*)d_in[0];
  const float* inw  = (const float*)d_in[1];
  const float* inb  = (const float*)d_in[2];
  const float* ow   = (const float*)d_in[3];
  const float* ob   = (const float*)d_in[4];
  const float* ln1w = (const float*)d_in[5];
  const float* ln1b = (const float*)d_in[6];
  const float* ln2w = (const float*)d_in[7];
  const float* ln2b = (const float*)d_in[8];
  const float* l1w  = (const float*)d_in[9];
  const float* l1b  = (const float*)d_in[10];
  const float* l2w  = (const float*)d_in[11];
  const float* l2b  = (const float*)d_in[12];
  const float* gw   = (const float*)d_in[13];
  const float* gb   = (const float*)d_in[14];
  float* out = (float*)d_out;

  dim3 grid(B_ * BPB);   // 2048 blocks, each loops over 8 chunks of 32 tokens
  dim3 block(256);
  mu_refine_kernel<<<grid, block, 0, stream>>>(mu, inw, inb, ow, ob, ln1w, ln1b,
                                               ln2w, ln2b, l1w, l1b, l2w, l2b,
                                               gw, gb, out);
}
